// SelfAttn_26620207300926
// MI455X (gfx1250) — matrix operand order repaired and checked by reading, not compiled, not run on hardware
//
#include <hip/hip_runtime.h>
#include <math.h>

typedef __bf16 bf16;
typedef __attribute__((ext_vector_type(16))) __bf16 bf16x16;
typedef __attribute__((ext_vector_type(8)))  __bf16 bf16x8;
typedef __attribute__((ext_vector_type(4)))  __bf16 bf16x4;
typedef __attribute__((ext_vector_type(8)))  float    f32x8;

#define DEV static __device__ __forceinline__

// ---- CDNA5 data-mover feature detection (compile-safe, layered) -----------
#if defined(__has_builtin)
#  if __has_builtin(__builtin_amdgcn_tensor_load_to_lds)
#    define HAVE_TDM 1
#  endif
#  if __has_builtin(__builtin_amdgcn_global_load_async_to_lds_b128)
#    define HAVE_ASYNC 1
#  endif
#endif
#ifndef HAVE_TDM
#  define HAVE_TDM 0
#endif
#ifndef HAVE_ASYNC
#  define HAVE_ASYNC 0
#endif
#if __has_include(<hip/amd_detail/amd_gfx1250_TDM.h>)
#  define TDM_SIX_ARGS 1          // therock-10.0 headers -> 6-arg builtin
#else
#  define TDM_SIX_ARGS 0          // ROCm 7.2 clang-22 -> 5-arg builtin
#endif

#if defined(__has_builtin) && __has_builtin(__builtin_amdgcn_s_wait_asynccnt)
#  define WAIT_ASYNC() __builtin_amdgcn_s_wait_asynccnt(0)
#else
#  define WAIT_ASYNC() asm volatile("s_wait_asynccnt 0x0" ::: "memory")
#endif

typedef unsigned int u32x4 __attribute__((ext_vector_type(4)));
typedef int          i32x4 __attribute__((ext_vector_type(4)));
typedef int          i32x8 __attribute__((ext_vector_type(8)));
typedef int          vi4   __attribute__((ext_vector_type(4)));

DEV unsigned lds_offset_of(const void* p) {
  return (unsigned)(__SIZE_TYPE__)(__attribute__((address_space(3))) const void*)p;
}

#if HAVE_TDM
// 1-D contiguous TDM copy global -> LDS (ndw dwords), per CDNA5 ISA D# layout.
DEV void tdm_load_1d(const void* gsrc, const void* ldsdst, unsigned ndw) {
  unsigned long long ga = (unsigned long long)(__SIZE_TYPE__)gsrc;
  u32x4 g0;
  g0[0] = 1u;                                            // count=1, no gather
  g0[1] = lds_offset_of(ldsdst);                         // lds_addr (bytes)
  g0[2] = (unsigned)(ga & 0xFFFFFFFFu);                  // global_addr[31:0]
  g0[3] = (unsigned)((ga >> 32) & 0x1FFFFFFu) | (2u << 30); // addr[56:32]|type=2
  i32x8 g1;
  g1[0] = (int)(2u << 16);                               // data_size=4B, mask=0
  g1[1] = (int)((ndw & 0xFFFFu) << 16);                  // tensor_dim0[15:0]
  g1[2] = (int)(((ndw >> 16) & 0xFFFFu) | (1u << 16));   // dim0[31:16] | dim1=1
  g1[3] = (int)((ndw & 0xFFFFu) << 16);                  // tile_dim0
  g1[4] = 1;                                             // tile_dim1=1
  g1[5] = (int)ndw;                                      // dim0_stride[31:0]
  g1[6] = 0;
  g1[7] = 0;
  i32x4 z4 = {0, 0, 0, 0};
#if TDM_SIX_ARGS
  i32x8 z8 = {0, 0, 0, 0, 0, 0, 0, 0};
  __builtin_amdgcn_tensor_load_to_lds(g0, g1, z4, z4, z8, 0);
#else
  __builtin_amdgcn_tensor_load_to_lds(g0, g1, z4, z4, 0);
#endif
}
#endif

#if HAVE_ASYNC
// per-lane async 16B copy global -> LDS (ASYNCcnt path)
DEV void async_copy16(const void* g, void* l) {
  __builtin_amdgcn_global_load_async_to_lds_b128(
      (__attribute__((address_space(1))) vi4*)(void*)g,
      (__attribute__((address_space(3))) vi4*)l, 0, 0);
}
#endif

DEV int lane_id() { return (int)(threadIdx.x & 31u); }

// round-to-nearest-even f32 -> bf16
DEV bf16 to_bf16(float f) {
  union { float f; unsigned u; } cv; cv.f = f;
  unsigned u = cv.u;
  unsigned r = u + 0x7FFFu + ((u >> 16) & 1u);
  union { unsigned short s; bf16 b; } out; out.s = (unsigned short)(r >> 16);
  return out.b;
}

// A-fragment 16x32 bf16 from row-major [16][ld] tile.
DEV bf16x16 load_afrag(const bf16* t, int ld) {
  int l = lane_id();
  const bf16* p = t + (l & 15) * ld + ((l >> 4) << 4);
  return *(const bf16x16*)p;
}

// B-fragment 32x16 bf16 from K-contiguous storage tile[n][k], leading dim ld.
DEV bf16x16 load_bfrag(const bf16* t, int ld) {
  int l = lane_id();
  const bf16* p = t + (l & 15) * ld + ((l >> 4) << 4);
  return *(const bf16x16*)p;
}

DEV f32x8 wmma_bf16(bf16x16 a, bf16x16 b, f32x8 c) {
  return __builtin_amdgcn_wmma_f32_16x16x32_bf16(false, a, false, b, (short)0, c, false, false);
}

#define QSCALE 0.17677669529663687f  // 1/sqrt(32), folded into q

// ---------------------------------------------------------------------------
// K1: y = BN(conv1x1(x, [w_qk;w_v]))  -> q/k/v bf16 in [ba][head][seq][hd]
// ---------------------------------------------------------------------------
__global__ __launch_bounds__(256) void k_proj_qkv(
    const float* __restrict__ x,
    const float* __restrict__ w_qk, const float* __restrict__ g_qk,
    const float* __restrict__ b_qk, const float* __restrict__ m_qk,
    const float* __restrict__ v_qk,
    const float* __restrict__ w_v,  const float* __restrict__ g_v,
    const float* __restrict__ b_v,  const float* __restrict__ m_v,
    const float* __restrict__ v_v,
    bf16* __restrict__ qbuf, bf16* __restrict__ kbuf, bf16* __restrict__ vbuf)
{
  __shared__ bf16 sW[64][256];   // [m][k]
  __shared__ bf16 sX[64][256];   // [n][k]
  __shared__ float sS[64];
  __shared__ float sT[64];

  const int b   = blockIdx.z;
  const int n0  = blockIdx.x * 64;
  const int o0  = blockIdx.y * 64;
  const int tid = threadIdx.x;

  if (tid < 64) {
    int o = o0 + tid;
    float g, bb, mm, vv;
    if (o < 512) { g = g_qk[o]; bb = b_qk[o]; mm = m_qk[o]; vv = v_qk[o]; }
    else { int oo = o - 512; g = g_v[oo]; bb = b_v[oo]; mm = m_v[oo]; vv = v_v[oo]; }
    float s = g * rsqrtf(vv + 1e-5f);
    sS[tid] = s; sT[tid] = bb - mm * s;
  }

  #pragma unroll
  for (int it = 0; it < 16; ++it) {              // weights: f32 -> bf16
    int idx = it * 256 + tid;
    int row = idx >> 6;
    int k4  = (idx & 63) << 2;
    int o = o0 + row;
    const float* wr = (o < 512) ? (w_qk + (size_t)o * 256)
                                : (w_v + (size_t)(o - 512) * 256);
    float4 w4 = *(const float4*)(wr + k4);
    bf16x4 pk;
    pk[0]=to_bf16(w4.x); pk[1]=to_bf16(w4.y); pk[2]=to_bf16(w4.z); pk[3]=to_bf16(w4.w);
    *(bf16x4*)&sW[row][k4] = pk;
  }
  #pragma unroll
  for (int it = 0; it < 16; ++it) {              // x tile: [c][n] -> sX[n][c]
    int idx = it * 256 + tid;
    int c  = idx >> 4;
    int n4 = (idx & 15) << 2;
    const float* xp = x + (((size_t)b * 256 + c) << 12) + n0 + n4;
    float4 x4 = *(const float4*)xp;
    sX[n4+0][c] = to_bf16(x4.x);
    sX[n4+1][c] = to_bf16(x4.y);
    sX[n4+2][c] = to_bf16(x4.z);
    sX[n4+3][c] = to_bf16(x4.w);
  }
  __syncthreads();

  const int wv = tid >> 5;
  const int mi = wv >> 1;
  const int ni = wv & 1;

  f32x8 acc0 = {0,0,0,0,0,0,0,0};
  f32x8 acc1 = {0,0,0,0,0,0,0,0};
  #pragma unroll
  for (int kb = 0; kb < 256; kb += 32) {
    bf16x16 a  = load_afrag(&sW[mi*16][kb], 256);
    bf16x16 b0 = load_bfrag(&sX[ni*32][kb], 256);
    bf16x16 b1 = load_bfrag(&sX[ni*32+16][kb], 256);
    acc0 = wmma_bf16(a, b0, acc0);
    acc1 = wmma_bf16(a, b1, acc1);
  }

  const int lane = tid & 31;
  const int col = lane & 15, hr = lane >> 4;
  const int orow = o0 + mi*16 + hr*8;
  const int sidx = mi*16 + hr*8;

  #pragma unroll
  for (int sub = 0; sub < 2; ++sub) {
    f32x8 acc = sub ? acc1 : acc0;
    int hw = n0 + ni*32 + sub*16 + col;
    int area = hw >> 10, ip = hw & 1023;
    int ba = b*4 + area;
    bf16* dst; float qs = 1.0f; int oc;
    if (orow < 256)      { oc = orow;       dst = qbuf; qs = QSCALE; }
    else if (orow < 512) { oc = orow - 256; dst = kbuf; }
    else                 { oc = orow - 512; dst = vbuf; }
    int head = oc >> 5, d0 = oc & 31;
    dst += ((size_t)((ba*8 + head)*1024 + ip) << 5) + d0;
    bf16x8 pk;
    #pragma unroll
    for (int r = 0; r < 8; ++r)
      pk[r] = to_bf16((acc[r] * sS[sidx + r] + sT[sidx + r]) * qs);
    *(bf16x8*)dst = pk;
  }
}

// ---------------------------------------------------------------------------
// K2: flash attention per (ba, head, 128-row block). K,V fully LDS-resident.
// K tile staged via async-to-LDS (or TDM when available).
// ---------------------------------------------------------------------------
__global__ __launch_bounds__(256) void k_attn(
    const bf16* __restrict__ qbuf, const bf16* __restrict__ kbuf,
    const bf16* __restrict__ vbuf, bf16* __restrict__ attnout)
{
  __shared__ bf16 sK[1024*32];       // [j][d]   64 KB
  __shared__ bf16 sVT[32*1024];      // [d][j]   64 KB
  __shared__ bf16 sP[8][16*32];      // per-wave P bounce tiles

  const int iblk = blockIdx.x, head = blockIdx.y, ba = blockIdx.z;
  const size_t hbase = (size_t)((ba*8 + head) * 1024) * 32;
  const bf16* kh = kbuf + hbase;
  const bf16* vh = vbuf + hbase;
  const bf16* qh = qbuf + hbase;
  const int tid = threadIdx.x;
  const int wv = tid >> 5, lane = tid & 31;

  __builtin_prefetch(qh, 0, 1);

  // ---- stage K tile: TDM > async-to-LDS > manual ----
#if HAVE_TDM
  if (wv == 0) {
    tdm_load_1d(kh, sK, 1024*32*2/4);          // 64 KB, one DMA descriptor
    __builtin_amdgcn_s_wait_tensorcnt(0);
  }
#elif HAVE_ASYNC
  #pragma unroll
  for (int it = 0; it < 16; ++it) {
    int c8 = it*256 + tid;
    async_copy16(kh + c8*8, sK + c8*8);
  }
#else
  #pragma unroll
  for (int it = 0; it < 16; ++it) {
    int c8 = it*256 + tid;
    *(bf16x8*)(sK + c8*8) = *(const bf16x8*)(kh + c8*8);
  }
#endif

  // ---- stage V transposed [j][d] -> [d][j] (needs VGPR pass) ----
  #pragma unroll
  for (int it = 0; it < 16; ++it) {
    int c8 = it*256 + tid;
    int j = c8 >> 2, d8 = (c8 & 3) << 3;
    bf16x8 vv = *(const bf16x8*)(vh + c8*8);
    #pragma unroll
    for (int e = 0; e < 8; ++e) sVT[(d8+e)*1024 + j] = vv[e];
  }
#if HAVE_ASYNC && !HAVE_TDM
  WAIT_ASYNC();
#endif
  __syncthreads();

  const int colj = lane & 15, hr = lane >> 4;
  const int i0 = iblk*128 + wv*16;

  bf16x16 aq = load_afrag(qh + (size_t)i0 * 32, 32);  // hd=32 == one K-step

  f32x8 o0 = {0,0,0,0,0,0,0,0};
  f32x8 o1 = {0,0,0,0,0,0,0,0};
  const f32x8 z = {0,0,0,0,0,0,0,0};
  float m[8], l[8];
  #pragma unroll
  for (int r = 0; r < 8; ++r) { m[r] = -INFINITY; l[r] = 0.0f; }

  bf16* pP = sP[wv];

  for (int j0 = 0; j0 < 1024; j0 += 32) {
    bf16x16 bk0 = load_bfrag(sK + j0*32, 32);
    bf16x16 bk1 = load_bfrag(sK + (j0+16)*32, 32);
    f32x8 s0 = wmma_bf16(aq, bk0, z);
    f32x8 s1 = wmma_bf16(aq, bk1, z);

    #pragma unroll
    for (int r = 0; r < 8; ++r) {                    // online softmax
      float a0 = s0[r], a1 = s1[r];
      float mx = fmaxf(a0, a1);
      mx = fmaxf(mx, __shfl_xor(mx, 1, 32));
      mx = fmaxf(mx, __shfl_xor(mx, 2, 32));
      mx = fmaxf(mx, __shfl_xor(mx, 4, 32));
      mx = fmaxf(mx, __shfl_xor(mx, 8, 32));
      float mn = fmaxf(m[r], mx);
      float al = __expf(m[r] - mn);
      float p0 = __expf(a0 - mn);
      float p1 = __expf(a1 - mn);
      float ps = p0 + p1;
      ps += __shfl_xor(ps, 1, 32);
      ps += __shfl_xor(ps, 2, 32);
      ps += __shfl_xor(ps, 4, 32);
      ps += __shfl_xor(ps, 8, 32);
      l[r] = l[r]*al + ps;
      m[r] = mn;
      o0[r] *= al; o1[r] *= al;
      int prow = hr*8 + r;
      pP[prow*32 + colj]      = to_bf16(p0);
      pP[prow*32 + 16 + colj] = to_bf16(p1);
    }
    __builtin_amdgcn_wave_barrier();
    bf16x16 ap  = load_afrag(pP, 32);
    bf16x16 bv0 = load_bfrag(sVT + j0, 1024);
    bf16x16 bv1 = load_bfrag(sVT + 16*1024 + j0, 1024);
    o0 = wmma_bf16(ap, bv0, o0);
    o1 = wmma_bf16(ap, bv1, o1);
    __builtin_amdgcn_wave_barrier();
  }

  const int bI = ba >> 2, area = ba & 3;
  #pragma unroll
  for (int r = 0; r < 8; ++r) {
    float inv = 1.0f / l[r];
    int ip = i0 + hr*8 + r;
    size_t p = ((size_t)bI*4096 + area*1024 + ip) * 256 + head*32;
    attnout[p + colj]      = to_bf16(o0[r] * inv);
    attnout[p + 16 + colj] = to_bf16(o1[r] * inv);
  }
}

// ---------------------------------------------------------------------------
// K3: out = BN(conv1x1(attnout, w_p)), f32 output in (B,C,H,W)
// ---------------------------------------------------------------------------
__global__ __launch_bounds__(256) void k_proj_out(
    const bf16* __restrict__ attnout,
    const float* __restrict__ w_p, const float* __restrict__ g_p,
    const float* __restrict__ b_p, const float* __restrict__ m_p,
    const float* __restrict__ v_p,
    float* __restrict__ out)
{
  __shared__ bf16 sW[64][256];
  __shared__ bf16 sX[64][256];
  __shared__ float sS[64];
  __shared__ float sT[64];

  const int p0  = blockIdx.x * 64;
  const int o0  = blockIdx.y * 64;
  const int tid = threadIdx.x;
  const int wv  = tid >> 5;

  // activations: straight bf16 copy -> data mover
#if HAVE_TDM
  if (wv == 0) {
    tdm_load_1d(attnout + (size_t)p0 * 256, &sX[0][0], 64*256*2/4); // 32 KB
    __builtin_amdgcn_s_wait_tensorcnt(0);
  }
#elif HAVE_ASYNC
  #pragma unroll
  for (int it = 0; it < 8; ++it) {
    int c8 = it*256 + tid;
    async_copy16(attnout + (size_t)p0*256 + c8*8, &sX[0][0] + c8*8);
  }
#else
  #pragma unroll
  for (int it = 0; it < 8; ++it) {
    int c8 = it*256 + tid;
    int row = c8 >> 5, off = (c8 & 31) << 3;
    *(bf16x8*)&sX[row][off] = *(const bf16x8*)(attnout + (size_t)(p0+row)*256 + off);
  }
#endif

  if (tid < 64) {
    int o = o0 + tid;
    float s = g_p[o] * rsqrtf(v_p[o] + 1e-5f);
    sS[tid] = s; sT[tid] = b_p[o] - m_p[o]*s;
  }
  #pragma unroll
  for (int it = 0; it < 16; ++it) {              // weights need f32->bf16
    int idx = it*256 + tid;
    int row = idx >> 6, k4 = (idx & 63) << 2;
    float4 w4 = *(const float4*)(w_p + (size_t)(o0+row)*256 + k4);
    bf16x4 pk;
    pk[0]=to_bf16(w4.x); pk[1]=to_bf16(w4.y); pk[2]=to_bf16(w4.z); pk[3]=to_bf16(w4.w);
    *(bf16x4*)&sW[row][k4] = pk;
  }
#if HAVE_ASYNC && !HAVE_TDM
  WAIT_ASYNC();
#endif
  __syncthreads();

  const int mi = wv >> 1, ni = wv & 1;
  f32x8 acc0 = {0,0,0,0,0,0,0,0};
  f32x8 acc1 = {0,0,0,0,0,0,0,0};
  #pragma unroll
  for (int kb = 0; kb < 256; kb += 32) {
    bf16x16 a  = load_afrag(&sW[mi*16][kb], 256);
    bf16x16 b0 = load_bfrag(&sX[ni*32][kb], 256);
    bf16x16 b1 = load_bfrag(&sX[ni*32+16][kb], 256);
    acc0 = wmma_bf16(a, b0, acc0);
    acc1 = wmma_bf16(a, b1, acc1);
  }
  const int lane = tid & 31, col = lane & 15, hr = lane >> 4;
  const int sidx = mi*16 + hr*8;
  const int bb = p0 >> 12;
  #pragma unroll
  for (int sub = 0; sub < 2; ++sub) {
    f32x8 acc = sub ? acc1 : acc0;
    int hw = (p0 & 4095) + ni*32 + sub*16 + col;
    #pragma unroll
    for (int r = 0; r < 8; ++r) {
      int o = o0 + sidx + r;
      out[((size_t)(bb*256 + o) << 12) + hw] = acc[r]*sS[sidx+r] + sT[sidx+r];
    }
  }
}

// ---------------------------------------------------------------------------
extern "C" void kernel_launch(void* const* d_in, const int* in_sizes, int n_in,
                              void* d_out, int out_size, void* d_ws, size_t ws_size,
                              hipStream_t stream) {
  (void)in_sizes; (void)n_in; (void)out_size; (void)ws_size;
  const float* x    = (const float*)d_in[0];
  const float* w_qk = (const float*)d_in[1];
  const float* g_qk = (const float*)d_in[2];
  const float* b_qk = (const float*)d_in[3];
  const float* m_qk = (const float*)d_in[4];
  const float* v_qk = (const float*)d_in[5];
  const float* w_v  = (const float*)d_in[6];
  const float* g_v  = (const float*)d_in[7];
  const float* b_v  = (const float*)d_in[8];
  const float* m_v  = (const float*)d_in[9];
  const float* v_v  = (const float*)d_in[10];
  const float* w_p  = (const float*)d_in[11];
  const float* g_p  = (const float*)d_in[12];
  const float* b_p  = (const float*)d_in[13];
  const float* m_p  = (const float*)d_in[14];
  const float* v_p  = (const float*)d_in[15];

  const size_t HSZ = (size_t)8 * 8 * 1024 * 32;   // q/k/v buffer elements
  bf16* q  = (bf16*)d_ws;
  bf16* k  = q + HSZ;
  bf16* v  = k + HSZ;
  bf16* ao = v + HSZ;                              // [8192][256] bf16

  k_proj_qkv<<<dim3(64, 12, 2), 256, 0, stream>>>(
      x, w_qk, g_qk, b_qk, m_qk, v_qk, w_v, g_v, b_v, m_v, v_v, q, k, v);
  k_attn<<<dim3(8, 8, 8), 256, 0, stream>>>(q, k, v, ao);
  k_proj_out<<<dim3(128, 4, 1), 256, 0, stream>>>(
      ao, w_p, g_p, b_p, m_p, v_p, (float*)d_out);
}
